// GS_71468255805590
// MI455X (gfx1250) — matrix operand order repaired and checked by reading, not compiled, not run on hardware
//
#include <hip/hip_runtime.h>
#include <hip/hip_bf16.h>

typedef __attribute__((ext_vector_type(16))) _Float16 v16h;
typedef __attribute__((ext_vector_type(8)))  float    v8f;

#define KPAD 1024   // K=1000 padded to multiple of 32 (zero-color pad gaussians)

// ---------------------------------------------------------------------------
// Prep kernel: clamp params, build inverse-covariance terms and f16 colors.
// g4[k]  = (mu_x, mu_y, ia, ib)     ia = inv[0][0], ib = inv[0][1]
// icv[k] = ic                        ic = inv[1][1]
// colT   = component-major f16 colors: colT[c*KPAD + k]
// Pad entries (k >= K): ia=ib=ic=0 -> alpha = exp(0) = 1, but color = 0,
// so padded gaussians contribute exactly 0 to the image.
// ---------------------------------------------------------------------------
__global__ void gs_prep_kernel(const float* __restrict__ mu,
                               const float* __restrict__ color,
                               const float* __restrict__ sxp,
                               const float* __restrict__ syp,
                               const float* __restrict__ rhop,
                               float4* __restrict__ g4,
                               float*  __restrict__ icv,
                               _Float16* __restrict__ colT,
                               int K) {
  int k = blockIdx.x * blockDim.x + threadIdx.x;
  if (k >= KPAD) return;
  if (k < K) {
    float sx = fminf(fmaxf(sxp[k], 1e-4f), 1.0f);
    float sy = fminf(fmaxf(syp[k], 1e-4f), 1.0f);
    float r  = fminf(fmaxf(rhop[k], -0.9f), 0.9f);
    float sx2 = sx * sx, sy2 = sy * sy, sxy = r * sx * sy;
    float inv = 1.0f / (sx2 * sy2 - sxy * sxy);   // det > 0 since |rho| <= 0.9
    g4[k]  = make_float4(mu[2 * k], mu[2 * k + 1], sy2 * inv, -sxy * inv);
    icv[k] = sx2 * inv;
    for (int c = 0; c < 3; ++c) {
      float cc = fminf(fmaxf(color[3 * k + c], 0.0f), 1.0f);
      colT[c * KPAD + k] = (_Float16)cc;
    }
  } else {
    g4[k]  = make_float4(0.0f, 0.0f, 0.0f, 0.0f);
    icv[k] = 0.0f;
    for (int c = 0; c < 3; ++c) colT[c * KPAD + k] = (_Float16)0.0f;
  }
}

// ---------------------------------------------------------------------------
// Main kernel: 8 waves per block, each wave owns a 16-pixel tile.
// Per 32-gaussian chunk: each lane computes 16 unique alphas (VALU + v_exp),
// converts to f16 into the WMMA A layout, loads its B column (consecutive
// halves -> ds_load_b128 x2), and issues v_wmma_f32_16x16x32_f16 to
// accumulate the 16x16 (16x3 used) image tile in f32.
// Lanes whose D-column is >= 3 load column 0's colors instead of zero:
// their accumulator columns are never stored, and this keeps the loop body
// fully convergent (no exec-mask manipulation around the WMMA).
// ---------------------------------------------------------------------------
__global__ __launch_bounds__(256) void gs_splat_kernel(
    const float*    __restrict__ pos,
    const float4*   __restrict__ g4,
    const float*    __restrict__ icv,
    const _Float16* __restrict__ colT,
    float* __restrict__ out,
    int N) {
  __shared__ __align__(16) float4   s_g[KPAD];        // 16 KB
  __shared__ __align__(16) float    s_ic[KPAD];       //  4 KB
  __shared__ __align__(32) _Float16 s_colT[3 * KPAD]; //  6 KB

  const int t = (int)threadIdx.x;
  for (int i = t; i < KPAD; i += 256) { s_g[i] = g4[i]; s_ic[i] = icv[i]; }
  for (int i = t; i < 3 * KPAD; i += 256) { s_colT[i] = colT[i]; }
  __syncthreads();

  const int wave = t >> 5;           // warpSize == 32 on gfx1250
  const int lane = t & 31;
  const int row  = lane & 15;        // A-matrix row M / B,D column N
  const int hi   = lane >> 4;        // which K-half this lane covers
  const int tile = (int)blockIdx.x * 8 + wave;

  int n = tile * 16 + row;
  int nc = n < N ? n : (N - 1);      // keep loads in-range w/o diverging EXEC
  const float px = pos[2 * nc];
  const float py = pos[2 * nc + 1];

  // A-layout K offsets for this lane (contiguous, matching B):
  //   e in [0,16): K = (hi?16:0) + e
  const int baseA = hi * 8;

  // B column pointer for this lane (16-bit B 32x16: lanes 0-15 K=e,
  // lanes 16-31 K=16+e, col=lane%16). Columns >= 3 alias column 0; those
  // accumulator columns are dead.
  const int bcol = (row < 3) ? row : 0;
  const _Float16* bsrc = &s_colT[bcol * KPAD + hi * 16];

  v8f acc = {};
#pragma unroll 2
  for (int k0 = 0; k0 < KPAD; k0 += 32) {
    // ---- build A: 16 alphas for pixel `row` at this lane's K slots ----
    v16h a;
#pragma unroll
    for (int e = 0; e < 16; ++e) {
      int kk = k0 + hi * 16 + e;
      float4 g = s_g[kk];
      float ic = s_ic[kk];
      float dx = px - g.x;
      float dy = py - g.y;
      float expo = -0.5f * (g.z * dx * dx + 2.0f * g.w * dx * dy + ic * dy * dy);
      a[e] = (_Float16)__expf(expo);
    }
    // ---- B: 16 consecutive halves of this lane's color column ----
    v16h b = *(const v16h*)(bsrc + k0);
    // D = A*B + C   (EXEC is all 1s here: no divergence around the WMMA)
    acc = __builtin_amdgcn_wmma_f32_16x16x32_f16(
        /*neg_a=*/false, a, /*neg_b=*/false, b,
        /*c_mod=*/(short)0, acc, /*reuse_a=*/false, /*reuse_b=*/false);
  }

  // ---- write out: C/D layout -> lane covers rows hi*8 .. hi*8+7 at col=row
  if (row < 3) {
#pragma unroll
    for (int r = 0; r < 8; ++r) {
      int m = tile * 16 + hi * 8 + r;
      if (m < N) {
        out[m * 3 + row] = tanhf(0.5f * acc[r]);
      }
    }
  }
}

extern "C" void kernel_launch(void* const* d_in, const int* in_sizes, int n_in,
                              void* d_out, int out_size, void* d_ws, size_t ws_size,
                              hipStream_t stream) {
  const float* pos   = (const float*)d_in[0];
  const float* mu    = (const float*)d_in[1];
  const float* color = (const float*)d_in[2];
  const float* sx    = (const float*)d_in[3];
  const float* sy    = (const float*)d_in[4];
  const float* rho   = (const float*)d_in[5];

  const int N = in_sizes[0] / 2;
  int K = in_sizes[3];
  if (K > KPAD) K = KPAD;

  char* ws = (char*)d_ws;
  float4*   g4   = (float4*)(ws);                          // 16 KB
  float*    icv  = (float*)(ws + sizeof(float4) * KPAD);   //  4 KB
  _Float16* colT = (_Float16*)(ws + sizeof(float4) * KPAD + sizeof(float) * KPAD);

  gs_prep_kernel<<<KPAD / 256, 256, 0, stream>>>(mu, color, sx, sy, rho,
                                                 g4, icv, colT, K);

  const int tiles  = (N + 15) / 16;
  const int blocks = (tiles + 7) / 8;
  gs_splat_kernel<<<blocks, 256, 0, stream>>>(pos, g4, icv, colT,
                                              (float*)d_out, N);
}
